// MFA_80960133530148
// MI455X (gfx1250) — compile-verified, matrix operand order checked by reading, not run on hardware
//
#include <hip/hip_runtime.h>
#include <hip/hip_bf16.h>
#include <math.h>

// MFA Woodbury log-likelihood, MI455X (gfx1250, wave32, WMMA).
// K=128 components, D=512 features, L=64 factors, N=4096 samples.
//
// All heavy contractions run on v_wmma_f32_16x16x32_bf16 with operands
// pre-packed into the exact CDNA5 WMMA fragment layouts (ISA 7.12.2), so hot
// loops are pure global_load_b128 + v_wmma. f32 accumulation throughout.

#define Nn 4096
#define Dd 512
#define Kc 128
#define Ld 64

typedef __bf16 v16bf __attribute__((ext_vector_type(16)));
typedef float  v8f   __attribute__((ext_vector_type(8)));

__device__ __forceinline__ unsigned short f32_to_bf16_rne(float f) {
  unsigned int u = __builtin_bit_cast(unsigned int, f);
  unsigned int r = u + 0x7FFFu + ((u >> 16) & 1u);
  return (unsigned short)(r >> 16);
}

// CDNA5 16-bit WMMA fragment element index (shared structure for A and B):
// lane = kh*16 + mc  (mc = row M for A-fragments / col N for B-fragments)
// VGPR v holds K-pair; dlocal = (v<4?0:16) + (v&3)*2 + kh*8 + lo.
// Returns flat index into the 512-element (32 lanes x 16 bf16) fragment.
__device__ __forceinline__ int frag_elem_index(int mc, int dlocal) {
  int hi  = dlocal >> 4;
  int rem = dlocal & 15;
  int kh  = rem >> 3;
  int r2  = rem & 7;
  int v   = (r2 >> 1) + (hi ? 4 : 0);
  int lo  = r2 & 1;
  return (kh * 16 + mc) * 16 + v * 2 + lo;
}

// ---------------------------------------------------------------------------
// Kernel 1: per-component precompute. One block per k.
//  - iD = D^-2, capacitance Lmat = I + A^T diag(iD) A (LDS-tiled)
//  - Gauss-Jordan inverse of SPD 64x64 in LDS (no pivoting) + log det
//  - pack AiD (B-frags), iL (B-frags), iD^T / -2(iD*MU)^T (B-frags) as bf16
//  - mut[l] = MU . AiD[:,l], muq = sum iD*MU^2, c_k = PI - 0.5*(cst+logdetSig)
// ---------------------------------------------------------------------------
__global__ __launch_bounds__(256) void mfa_precomp(
    const float* __restrict__ MU, const float* __restrict__ A,
    const float* __restrict__ Dm, const float* __restrict__ PI,
    unsigned short* __restrict__ AiDpack, unsigned short* __restrict__ iLpack,
    unsigned short* __restrict__ iDTpack, unsigned short* __restrict__ bTpack,
    float* __restrict__ mut, float* __restrict__ cK, float* __restrict__ muq) {
  const int k = blockIdx.x, tid = threadIdx.x;
  __shared__ float iD_s[Dd];
  __shared__ float chunk[64][65];      // 64 d-rows x 64 l-cols of A (padded)
  __shared__ float aug[64][130];       // [Lmat | I] for Gauss-Jordan
  __shared__ float red[256];
  __shared__ float colp[64];
  __shared__ float s_logdet;

  const float* Ak  = A  + (size_t)k * Dd * Ld;
  const float* MUk = MU + (size_t)k * Dd;
  const float* Dk  = Dm + (size_t)k * Dd;

  for (int d = tid; d < Dd; d += 256) {
    float dv = Dk[d];
    iD_s[d] = 1.0f / (dv * dv);
  }
  __syncthreads();

  // reductions: sum log(iD), sum iD*MU^2
  float p_log = 0.f, p_muq = 0.f;
  for (int d = tid; d < Dd; d += 256) {
    float id = iD_s[d], mu = MUk[d];
    p_log += __logf(id);
    p_muq += id * mu * mu;
  }
  red[tid] = p_log; __syncthreads();
  for (int s = 128; s > 0; s >>= 1) { if (tid < s) red[tid] += red[tid + s]; __syncthreads(); }
  float sumlogiD = red[0]; __syncthreads();
  red[tid] = p_muq; __syncthreads();
  for (int s = 128; s > 0; s >>= 1) { if (tid < s) red[tid] += red[tid + s]; __syncthreads(); }
  float muqv = red[0]; __syncthreads();

  // Lmat[l][m] = sum_d A[d][l]*A[d][m]*iD[d], LDS-tiled over d
  const int l  = tid >> 2;
  const int m0 = (tid & 3) * 16;
  float acc[16];
#pragma unroll
  for (int j = 0; j < 16; ++j) acc[j] = 0.f;
  for (int c0 = 0; c0 < Dd; c0 += 64) {
    for (int i = tid; i < 64 * 64; i += 256) {
      int dd = i >> 6, ll = i & 63;
      chunk[dd][ll] = Ak[(size_t)(c0 + dd) * Ld + ll];
    }
    __syncthreads();
    for (int dd = 0; dd < 64; ++dd) {
      float ail = chunk[dd][l] * iD_s[c0 + dd];
#pragma unroll
      for (int j = 0; j < 16; ++j) acc[j] += ail * chunk[dd][m0 + j];
    }
    __syncthreads();
  }
#pragma unroll
  for (int j = 0; j < 16; ++j) {
    int m = m0 + j;
    aug[l][m]      = acc[j] + (l == m ? 1.0f : 0.0f);
    aug[l][64 + m] = (l == m) ? 1.0f : 0.0f;
  }
  if (tid == 0) s_logdet = 0.f;
  __syncthreads();

  // Gauss-Jordan (SPD: no pivoting); log det = sum log(pivots)
  for (int p = 0; p < 64; ++p) {
    float piv = aug[p][p];
    __syncthreads();
    if (tid == 0) s_logdet += __logf(piv);
    float rp = 1.0f / piv;
    if (tid < 128) aug[p][tid] *= rp;
    __syncthreads();
    if (tid < 64) colp[tid] = aug[tid][p];
    __syncthreads();
    for (int i = tid; i < 64 * 128; i += 256) {
      int r = i >> 7, c = i & 127;
      if (r != p) aug[r][c] -= colp[r] * aug[p][c];
    }
    __syncthreads();
  }

  // mut[l] = sum_d MU[d]*iD[d]*A[d][l]
  if (tid < Ld) {
    float s = 0.f;
    for (int d = 0; d < Dd; ++d) s += MUk[d] * iD_s[d] * Ak[(size_t)d * Ld + tid];
    mut[k * Ld + tid] = s;
  }

  // pack iL as bf16 B-fragments: [k][kb(2)][mb(4)][512]
  for (int i = tid; i < 64 * 64; i += 256) {
    int lr = i >> 6, mc = i & 63;
    int kb = lr >> 5, dl = lr & 31;
    int mb = mc >> 4, c = mc & 15;
    size_t frag = (size_t)(k * 2 + kb) * 4 + mb;
    iLpack[frag * 512 + frag_elem_index(c, dl)] = f32_to_bf16_rne(aug[lr][64 + mc]);
  }

  // pack AiD = diag(iD)*A as bf16 B-fragments: [k][dblk(16)][lblk(4)][512]
  for (int i = tid; i < Dd * Ld; i += 256) {
    int d = i >> 6, ll = i & 63;
    int dblk = d >> 5, dl = d & 31;
    int lblk = ll >> 4, c = ll & 15;
    size_t frag = (size_t)((k * 16 + dblk) * 4 + lblk);
    AiDpack[frag * 512 + frag_elem_index(c, dl)] =
        f32_to_bf16_rne(Ak[(size_t)d * Ld + ll] * iD_s[d]);
  }

  // pack iD^T and -2*(iD*MU)^T as B-fragments (D x K): [dblk(16)][kblk(8)][512]
  for (int d = tid; d < Dd; d += 256) {
    int dblk = d >> 5, dl = d & 31;
    int kblk = k >> 4, c = k & 15;
    size_t idx = ((size_t)(dblk * 8 + kblk)) * 512 + frag_elem_index(c, dl);
    float id = iD_s[d];
    iDTpack[idx] = f32_to_bf16_rne(id);
    bTpack[idx]  = f32_to_bf16_rne(-2.0f * id * MUk[d]);
  }

  if (tid == 0) {
    muq[k] = muqv;
    const float LOG2PI = 1.8378770664093453f;
    cK[k] = PI[k] - 0.5f * ((float)Dd * LOG2PI + s_logdet - sumlogiD);
  }
}

// ---------------------------------------------------------------------------
// Kernel 2: pack x and x^2 into bf16 WMMA A-fragments: [ntile(256)][dblk(16)]
// ---------------------------------------------------------------------------
__global__ __launch_bounds__(32) void mfa_pack_x(
    const float* __restrict__ x, unsigned short* __restrict__ xpack,
    unsigned short* __restrict__ x2pack) {
  int ntile = blockIdx.x, dblk = blockIdx.y, lane = threadIdx.x;
  int m = lane & 15, kh = lane >> 4;
  int n = ntile * 16 + m;
  size_t base = ((size_t)ntile * 16 + dblk) * 512 + lane * 16;
#pragma unroll
  for (int e = 0; e < 16; ++e) {
    int v = e >> 1, lo = e & 1;
    int dl = ((v < 4) ? 0 : 16) + (v & 3) * 2 + kh * 8 + lo;
    float val = x[(size_t)n * Dd + dblk * 32 + dl];
    xpack[base + e]  = f32_to_bf16_rne(val);
    x2pack[base + e] = f32_to_bf16_rne(val * val);
  }
}

// ---------------------------------------------------------------------------
// Kernel 3: q1 = X^2 @ iD^T + X @ (-2 iD MU)^T + muq -> pc[n][k]
// One wave per 16x16 output tile; two WMMAs per d-block.
// ---------------------------------------------------------------------------
__global__ __launch_bounds__(32) void mfa_q1(
    const unsigned short* __restrict__ xpack, const unsigned short* __restrict__ x2pack,
    const unsigned short* __restrict__ iDTpack, const unsigned short* __restrict__ bTpack,
    const float* __restrict__ muq, float* __restrict__ pc) {
  int ntile = blockIdx.x, kblk = blockIdx.y, lane = threadIdx.x;
  v8f acc;
#pragma unroll
  for (int r = 0; r < 8; ++r) acc[r] = 0.f;
  for (int dblk = 0; dblk < 16; ++dblk) {
    size_t abase = ((size_t)ntile * 16 + dblk) * 512 + lane * 16;
    size_t bbase = ((size_t)(dblk * 8 + kblk)) * 512 + lane * 16;
    v16bf a2 = *(const v16bf*)(x2pack + abase);
    v16bf bD = *(const v16bf*)(iDTpack + bbase);
    acc = __builtin_amdgcn_wmma_f32_16x16x32_bf16(false, a2, false, bD, (short)0, acc, false, false);
    v16bf a1 = *(const v16bf*)(xpack + abase);
    v16bf bB = *(const v16bf*)(bTpack + bbase);
    acc = __builtin_amdgcn_wmma_f32_16x16x32_bf16(false, a1, false, bB, (short)0, acc, false, false);
  }
  int c = lane & 15, half = lane >> 4;
  int kcol = kblk * 16 + c;
  float mq = muq[kcol];
#pragma unroll
  for (int r = 0; r < 8; ++r) {
    int n = ntile * 16 + r + half * 8;
    pc[(size_t)n * Kc + kcol] = acc[r] + mq;
  }
}

// ---------------------------------------------------------------------------
// Kernel 4: main Woodbury kernel. One wave per (16-row n-tile, component k).
//  t = X@AiD - mut (4 WMMA accs, 16-step K-loop over D=512)
//  LDS transpose t -> bf16 A-frags; U = t @ iL (8 WMMAs)
//  q2[n] = sum_l t*U (lane-group shuffle reduce); pc = cK - q1/2 + q2/2
// ---------------------------------------------------------------------------
__global__ __launch_bounds__(32) void mfa_main(
    const unsigned short* __restrict__ xpack, const unsigned short* __restrict__ AiDpack,
    const unsigned short* __restrict__ iLpack, const float* __restrict__ mut,
    const float* __restrict__ cK, float* __restrict__ pc) {
  int ntile = blockIdx.x, k = blockIdx.y, lane = threadIdx.x;
  __shared__ float t_lds[16][68];

  v8f acc[4];
#pragma unroll
  for (int j = 0; j < 4; ++j)
#pragma unroll
    for (int r = 0; r < 8; ++r) acc[j][r] = 0.f;

  for (int dblk = 0; dblk < 16; ++dblk) {
    size_t abase = ((size_t)ntile * 16 + dblk) * 512 + lane * 16;
    v16bf a = *(const v16bf*)(xpack + abase);
    if (dblk < 15)  // prefetch next x fragment (global_prefetch_b8)
      __builtin_prefetch(xpack + abase + 512, 0, 1);
#pragma unroll
    for (int j = 0; j < 4; ++j) {
      v16bf b = *(const v16bf*)(AiDpack + ((size_t)((k * 16 + dblk) * 4 + j)) * 512 + lane * 16);
      acc[j] = __builtin_amdgcn_wmma_f32_16x16x32_bf16(false, a, false, b, (short)0, acc[j], false, false);
    }
  }

  int c = lane & 15, half = lane >> 4;
#pragma unroll
  for (int j = 0; j < 4; ++j) {
    float mval = mut[k * Ld + j * 16 + c];
#pragma unroll
    for (int r = 0; r < 8; ++r) {
      acc[j][r] -= mval;
      t_lds[r + half * 8][j * 16 + c] = acc[j][r];
    }
  }
  __syncthreads();

  // re-feed t as bf16 A-fragments (two 32-wide contraction blocks)
  v16bf ta[2];
  int mm = lane & 15, kh = lane >> 4;
#pragma unroll
  for (int kb = 0; kb < 2; ++kb)
#pragma unroll
    for (int e = 0; e < 16; ++e) {
      int v = e >> 1, lo = e & 1;
      int dl = ((v < 4) ? 0 : 16) + (v & 3) * 2 + kh * 8 + lo;
      ta[kb][e] = __builtin_bit_cast(__bf16, f32_to_bf16_rne(t_lds[mm][kb * 32 + dl]));
    }

  v8f U[4];
#pragma unroll
  for (int j = 0; j < 4; ++j)
#pragma unroll
    for (int r = 0; r < 8; ++r) U[j][r] = 0.f;
#pragma unroll
  for (int kb = 0; kb < 2; ++kb)
#pragma unroll
    for (int j = 0; j < 4; ++j) {
      v16bf b = *(const v16bf*)(iLpack + ((size_t)((k * 2 + kb) * 4 + j)) * 512 + lane * 16);
      U[j] = __builtin_amdgcn_wmma_f32_16x16x32_bf16(false, ta[kb], false, b, (short)0, U[j], false, false);
    }

  float q2v[8];
#pragma unroll
  for (int r = 0; r < 8; ++r) {
    float s = 0.f;
#pragma unroll
    for (int j = 0; j < 4; ++j) s += acc[j][r] * U[j][r];
#pragma unroll
    for (int off = 8; off > 0; off >>= 1) s += __shfl_xor(s, off, 16);
    q2v[r] = s;
  }

  if (c == 0) {
    float ckv = cK[k];
#pragma unroll
    for (int r = 0; r < 8; ++r) {
      int n = ntile * 16 + r + half * 8;
      size_t idx = (size_t)n * Kc + k;
      pc[idx] = ckv - 0.5f * pc[idx] + 0.5f * q2v[r];
    }
  }
}

// ---------------------------------------------------------------------------
// Kernel 5: logsumexp over K=128 per row n
// ---------------------------------------------------------------------------
__global__ __launch_bounds__(256) void mfa_lse(const float* __restrict__ pc,
                                               float* __restrict__ out) {
  int n = blockIdx.x * 256 + threadIdx.x;
  const float* row = pc + (size_t)n * Kc;
  float mx = -INFINITY;
  for (int kk = 0; kk < Kc; ++kk) mx = fmaxf(mx, row[kk]);
  float s = 0.f;
  for (int kk = 0; kk < Kc; ++kk) s += __expf(row[kk] - mx);
  out[n] = mx + __logf(s);
}

// ---------------------------------------------------------------------------
extern "C" void kernel_launch(void* const* d_in, const int* in_sizes, int n_in,
                              void* d_out, int out_size, void* d_ws, size_t ws_size,
                              hipStream_t stream) {
  const float* x  = (const float*)d_in[0];
  const float* MU = (const float*)d_in[1];
  const float* A  = (const float*)d_in[2];
  const float* Dm = (const float*)d_in[3];
  const float* PI = (const float*)d_in[4];
  float* out = (float*)d_out;

  // workspace layout (~19.3 MB total)
  char* ws = (char*)d_ws;
  unsigned short* xpack   = (unsigned short*)(ws);                              // 4 MB
  unsigned short* x2pack  = (unsigned short*)(ws + (size_t)(4 << 20));          // 4 MB
  unsigned short* AiDpack = (unsigned short*)(ws + (size_t)(8 << 20));          // 8 MB
  unsigned short* iLpack  = (unsigned short*)(ws + (size_t)(16 << 20));         // 1 MB
  unsigned short* iDTpack = (unsigned short*)(ws + (size_t)(17 << 20));         // 128 KB
  unsigned short* bTpack  = (unsigned short*)(ws + (size_t)(17 << 20) + (128 << 10));
  float* mut = (float*)(ws + (size_t)(17 << 20) + (256 << 10));                 // 32 KB
  float* cK  = (float*)(ws + (size_t)(17 << 20) + (288 << 10));
  float* muq = (float*)(ws + (size_t)(17 << 20) + (292 << 10));
  float* pc  = (float*)(ws + (size_t)(17 << 20) + (296 << 10));                 // 2 MB

  mfa_precomp<<<Kc, 256, 0, stream>>>(MU, A, Dm, PI, AiDpack, iLpack, iDTpack,
                                      bTpack, mut, cK, muq);
  mfa_pack_x<<<dim3(Nn / 16, Dd / 32), 32, 0, stream>>>(x, xpack, x2pack);
  mfa_q1<<<dim3(Nn / 16, Kc / 16), 32, 0, stream>>>(xpack, x2pack, iDTpack,
                                                    bTpack, muq, pc);
  mfa_main<<<dim3(Nn / 16, Kc), 32, 0, stream>>>(xpack, AiDpack, iLpack, mut, cK, pc);
  mfa_lse<<<Nn / 256, 256, 0, stream>>>(pc, out);
}
